// Qwen3MoeMoE_63952063038001
// MI455X (gfx1250) — matrix operand order repaired and checked by reading, not compiled, not run on hardware
//
#include <hip/hip_runtime.h>
#include <hip/hip_bf16.h>
#include <cstddef>
#include <cstdint>

// Problem dims (match reference)
#define TOK   4096          // B*S tokens
#define HDIM  2048          // hidden
#define IDIM  768           // moe intermediate
#define NEXP  64            // experts
#define TOPK  8
#define NROWS (TOK*TOPK)    // 32768 expanded rows

// GEMM tiling
#define BM 128
#define BN 128
#define BK 32
#define APAD 40             // padded k-stride in shorts (16B-aligned fragments, fewer bank conflicts)

typedef __attribute__((ext_vector_type(16))) __bf16        v16bf;
typedef __attribute__((ext_vector_type(8)))  float         v8f;
typedef __attribute__((ext_vector_type(4)))  unsigned int  v4u;
typedef __attribute__((ext_vector_type(8)))  int           v8i;
typedef __attribute__((ext_vector_type(4)))  int           v4i;

#if defined(__has_builtin)
# if __has_builtin(__builtin_amdgcn_tensor_load_to_lds) && __has_builtin(__builtin_amdgcn_s_wait_tensorcnt)
#  define HAVE_TDM 1
# endif
#endif
#ifndef HAVE_TDM
# define HAVE_TDM 0
#endif

union Frag {
    v16bf v;
    uint4 q[2];
};

__device__ __forceinline__ unsigned short f2bf(float f) {
    unsigned int u = __float_as_uint(f);
    u += 0x7FFFu + ((u >> 16) & 1u);   // round-to-nearest-even
    return (unsigned short)(u >> 16);
}

#if HAVE_TDM
// Issue a TDM 2D tile load: [128 rows x 32 bf16] from a row-major bf16 tensor
// (row stride IDIM elements) into LDS with a 40-short padded row stride.
// tensor_dim1 = rowsRem so rows past the group's end read as zero.
__device__ __forceinline__ void tdm_load_a_tile(const unsigned short* gptr,
                                                unsigned lds_addr, int rowsRem) {
    const unsigned long long ga = (unsigned long long)(uintptr_t)gptr;
    v4u g0;
    g0[0] = 1u;                                            // count=1, user mode
    g0[1] = lds_addr;                                      // lds_addr [63:32]
    g0[2] = (unsigned)(ga & 0xFFFFFFFFull);                // global_addr lo
    g0[3] = (unsigned)((ga >> 32) & 0x1FFFFFFull)          // global_addr[56:32]
          | (2u << 30);                                    // type=2 ("image")
    v8i g1;
    // data_size=1(2B), pad_enable, pad_interval=3(16 dwords), pad_amount=3(4 dwords)
    g1[0] = (int)((1u << 16) | (1u << 20) | (3u << 22) | (3u << 25));
    g1[1] = (int)(((unsigned)IDIM & 0xFFFFu) << 16);       // tensor_dim0 lo16 (abar addr=0)
    g1[2] = (int)((((unsigned)IDIM >> 16) & 0xFFFFu)       // tensor_dim0 hi16
          | (((unsigned)rowsRem & 0xFFFFu) << 16));        // tensor_dim1 lo16
    g1[3] = (int)((((unsigned)rowsRem >> 16) & 0xFFFFu)    // tensor_dim1 hi16
          | (32u << 16));                                  // tile_dim0 = 32
    g1[4] = (int)128u;                                     // tile_dim1 = 128, tile_dim2 = 0
    g1[5] = (int)IDIM;                                     // tensor_dim0_stride lo32
    g1[6] = 0;                                             // stride hi16 / dim1_stride lo16
    g1[7] = 0;
    const v4i z4 = {0, 0, 0, 0};
    const v8i z8 = {0, 0, 0, 0, 0, 0, 0, 0};
    // 6-arg toolchain form: (g0, g1, g2, g3, g4, cpol)
    __builtin_amdgcn_tensor_load_to_lds(g0, g1, z4, z4, z8, 0);
}
#endif

// ---------------------------------------------------------------- router GEMM
__global__ __launch_bounds__(256)
void router_gemm(const float* __restrict__ x, const float* __restrict__ gw,
                 float* __restrict__ logits) {
    __shared__ float xs[32][65];
    __shared__ float wt[64][65];
    const int tid  = threadIdx.x;
    const int tok0 = blockIdx.x * 32;
    const int tl   = tid & 31;
    const int eb   = (tid >> 5) * 8;

    float acc[8];
#pragma unroll
    for (int j = 0; j < 8; ++j) acc[j] = 0.f;

    for (int hc = 0; hc < HDIM / 64; ++hc) {
        const int h0 = hc * 64;
        {
            const int i = tid * 8, r = i >> 6, c = i & 63;
            const float4* src = (const float4*)(x + (size_t)(tok0 + r) * HDIM + h0 + c);
            float4 a = src[0], b = src[1];
            xs[r][c + 0] = a.x; xs[r][c + 1] = a.y; xs[r][c + 2] = a.z; xs[r][c + 3] = a.w;
            xs[r][c + 4] = b.x; xs[r][c + 5] = b.y; xs[r][c + 6] = b.z; xs[r][c + 7] = b.w;
        }
        {
            const int r = tid >> 2, c = (tid & 3) * 16;
            const float4* src = (const float4*)(gw + (size_t)r * HDIM + h0 + c);
#pragma unroll
            for (int q = 0; q < 4; ++q) {
                float4 v = src[q];
                wt[r][c + q * 4 + 0] = v.x; wt[r][c + q * 4 + 1] = v.y;
                wt[r][c + q * 4 + 2] = v.z; wt[r][c + q * 4 + 3] = v.w;
            }
        }
        __syncthreads();
        for (int k = 0; k < 64; ++k) {
            const float xv = xs[tl][k];
#pragma unroll
            for (int j = 0; j < 8; ++j) acc[j] = fmaf(xv, wt[eb + j][k], acc[j]);
        }
        __syncthreads();
    }
#pragma unroll
    for (int j = 0; j < 8; ++j)
        logits[(size_t)(tok0 + tl) * NEXP + eb + j] = acc[j];
}

// ------------------------------------------------- softmax + top-8 + histogram
__global__ void zero_counts(int* __restrict__ counts) {
    if (threadIdx.x < NEXP) counts[threadIdx.x] = 0;
}

__global__ void topk_kernel(const float* __restrict__ logits,
                            float* __restrict__ topkw, int* __restrict__ topki,
                            int* __restrict__ counts) {
    const int t = blockIdx.x * blockDim.x + threadIdx.x;
    if (t >= TOK) return;
    const float* l = logits + (size_t)t * NEXP;
    float sc[NEXP];
    float m = -3.4e38f;
    for (int e = 0; e < NEXP; ++e) m = fmaxf(m, l[e]);
    for (int e = 0; e < NEXP; ++e) sc[e] = __expf(l[e] - m);
    float wsum = 0.f;
    float selv[TOPK]; int seli[TOPK];
    for (int j = 0; j < TOPK; ++j) {
        float best = -1.f; int bi = 0;
        for (int e = 0; e < NEXP; ++e)
            if (sc[e] > best) { best = sc[e]; bi = e; }
        selv[j] = best; seli[j] = bi; sc[bi] = -2.f; wsum += best;
    }
    const float inv = 1.f / wsum;
    for (int j = 0; j < TOPK; ++j) {
        topkw[t * TOPK + j] = selv[j] * inv;
        topki[t * TOPK + j] = seli[j];
        atomicAdd(&counts[seli[j]], 1);
    }
}

__global__ void offsets_kernel(const int* __restrict__ counts,
                               int* __restrict__ offs, int* __restrict__ fill) {
    if (threadIdx.x == 0 && blockIdx.x == 0) {
        int off = 0;
        for (int e = 0; e < NEXP; ++e) { offs[e] = off; fill[e] = off; off += counts[e]; }
        offs[NEXP] = off;
    }
}

__global__ void scatter_kernel(const int* __restrict__ topki, int* __restrict__ fill,
                               int* __restrict__ perm, int* __restrict__ inv) {
    const int i = blockIdx.x * blockDim.x + threadIdx.x;
    if (i >= NROWS) return;
    const int e = topki[i];
    const int slot = atomicAdd(&fill[e], 1);
    perm[slot] = i / TOPK;
    inv[i] = slot;
}

// --------------------------------------- fused gate+up WMMA GEMM, SiLU epilogue
// Double-buffered LDS: global loads for tile kt+1 overlap WMMA of tile kt.
__global__ __launch_bounds__(256)
void gateup_gemm(const float* __restrict__ x, const float* __restrict__ wg,
                 const float* __restrict__ wu, const int* __restrict__ perm,
                 const int* __restrict__ offs, unsigned short* __restrict__ hbuf) {
    __shared__ unsigned short As[2][BM][APAD];
    __shared__ unsigned short Bgs[2][BN][APAD];
    __shared__ unsigned short Bus[2][BN][APAD];

    const int e    = blockIdx.z;
    const int row0 = offs[e];
    const int Me   = offs[e + 1] - row0;
    const int m0   = blockIdx.y * BM;
    if (m0 >= Me) return;
    const int n0   = blockIdx.x * BN;

    const int tid  = threadIdx.x;
    const int lane = tid & 31, wave = tid >> 5;
    const int half = lane >> 4, ln = lane & 15;

    const v8f vzero = {};
    v8f accG[8], accU[8];
#pragma unroll
    for (int i = 0; i < 8; ++i) { accG[i] = vzero; accU[i] = vzero; }

    const int arow = tid >> 1;
    const int acol = (tid & 1) * 16;
    int tokenRow = -1;
    if (m0 + arow < Me) tokenRow = perm[row0 + m0 + arow];

    const int bk = tid >> 3;            // 0..31
    const int bc = (tid & 7) * 16;      // 0..112
    const size_t wbase = (size_t)e * HDIM * IDIM;

    unsigned apk[8];                    // A prefetch, packed bf16 pairs
    float4 bg[4], bu[4];                // B prefetch

    auto loadGlobal = [&](int kt) {
        const int k0 = kt * BK;
        if (tokenRow >= 0) {
            const float4* src = (const float4*)(x + (size_t)tokenRow * HDIM + k0 + acol);
#pragma unroll
            for (int q = 0; q < 4; ++q) {
                float4 v = src[q];
                apk[q * 2 + 0] = (unsigned)f2bf(v.x) | ((unsigned)f2bf(v.y) << 16);
                apk[q * 2 + 1] = (unsigned)f2bf(v.z) | ((unsigned)f2bf(v.w) << 16);
            }
        } else {
#pragma unroll
            for (int q = 0; q < 8; ++q) apk[q] = 0u;
        }
        const size_t roff = wbase + (size_t)(k0 + bk) * IDIM + n0 + bc;
        const float4* sg = (const float4*)(wg + roff);
        const float4* su = (const float4*)(wu + roff);
#pragma unroll
        for (int q = 0; q < 4; ++q) { bg[q] = sg[q]; bu[q] = su[q]; }
    };
    auto storeLds = [&](int buf) {
#pragma unroll
        for (int q = 0; q < 8; ++q)
            *(unsigned*)&As[buf][arow][acol + q * 2] = apk[q];
#pragma unroll
        for (int q = 0; q < 4; ++q) {
            Bgs[buf][bc + q * 4 + 0][bk] = f2bf(bg[q].x);
            Bgs[buf][bc + q * 4 + 1][bk] = f2bf(bg[q].y);
            Bgs[buf][bc + q * 4 + 2][bk] = f2bf(bg[q].z);
            Bgs[buf][bc + q * 4 + 3][bk] = f2bf(bg[q].w);
            Bus[buf][bc + q * 4 + 0][bk] = f2bf(bu[q].x);
            Bus[buf][bc + q * 4 + 1][bk] = f2bf(bu[q].y);
            Bus[buf][bc + q * 4 + 2][bk] = f2bf(bu[q].z);
            Bus[buf][bc + q * 4 + 3][bk] = f2bf(bu[q].w);
        }
    };

    loadGlobal(0);
    storeLds(0);

    const int NK = HDIM / BK;
    for (int kt = 0; kt < NK; ++kt) {
        __syncthreads();
        const int cur = kt & 1;
        const bool more = (kt + 1 < NK);
        if (more) loadGlobal(kt + 1);   // HBM loads overlap WMMA below

        Frag fa;
        fa.q[0] = *(const uint4*)&As[cur][wave * 16 + ln][half * 8];
        fa.q[1] = *(const uint4*)&As[cur][wave * 16 + ln][16 + half * 8];
        Frag fg, fu;
        fg.q[0] = *(const uint4*)&Bgs[cur][ln][half * 8];
        fg.q[1] = *(const uint4*)&Bgs[cur][ln][16 + half * 8];
        fu.q[0] = *(const uint4*)&Bus[cur][ln][half * 8];
        fu.q[1] = *(const uint4*)&Bus[cur][ln][16 + half * 8];
#pragma unroll
        for (int ns = 0; ns < 8; ++ns) {
            Frag fgn, fun;
            if (ns < 7) {               // pipeline next fragments ahead of the WMMAs
                fgn.q[0] = *(const uint4*)&Bgs[cur][(ns + 1) * 16 + ln][half * 8];
                fgn.q[1] = *(const uint4*)&Bgs[cur][(ns + 1) * 16 + ln][16 + half * 8];
                fun.q[0] = *(const uint4*)&Bus[cur][(ns + 1) * 16 + ln][half * 8];
                fun.q[1] = *(const uint4*)&Bus[cur][(ns + 1) * 16 + ln][16 + half * 8];
            } else { fgn = fg; fun = fu; }
            accG[ns] = __builtin_amdgcn_wmma_f32_16x16x32_bf16(
                false, fa.v, false, fg.v, (short)0, accG[ns], false, false);
            accU[ns] = __builtin_amdgcn_wmma_f32_16x16x32_bf16(
                false, fa.v, false, fu.v, (short)0, accU[ns], false, false);
            fg = fgn; fu = fun;
        }
        if (more) storeLds(cur ^ 1);
    }
    // epilogue: h = silu(g) * u -> bf16
#pragma unroll
    for (int ns = 0; ns < 8; ++ns) {
#pragma unroll
        for (int r = 0; r < 8; ++r) {
            const int gm = m0 + wave * 16 + half * 8 + r;
            if (gm < Me) {
                const float g = accG[ns][r], u = accU[ns][r];
                const float h = (g / (1.f + __expf(-g))) * u;
                hbuf[(size_t)(row0 + gm) * IDIM + n0 + ns * 16 + ln] = f2bf(h);
            }
        }
    }
}

// ------------------------------------------------------- down WMMA GEMM
// A tile is a contiguous bf16 block -> use the Tensor Data Mover when available;
// the TDM DMA overlaps the manual f32->bf16 B-tile staging.
__global__ __launch_bounds__(256)
void down_gemm(const unsigned short* __restrict__ hbuf, const float* __restrict__ wd,
               const int* __restrict__ offs, float* __restrict__ outbuf) {
    __shared__ unsigned short As[BM][APAD];
    __shared__ unsigned short Bs[BN][APAD];

    const int e    = blockIdx.z;
    const int row0 = offs[e];
    const int Me   = offs[e + 1] - row0;
    const int m0   = blockIdx.y * BM;
    if (m0 >= Me) return;
    const int n0   = blockIdx.x * BN;

    const int tid  = threadIdx.x;
    const int lane = tid & 31, wave = tid >> 5;
    const int half = lane >> 4, ln = lane & 15;
    const int rowsRem = Me - m0;

    const v8f vzero = {};
    v8f acc[8];
#pragma unroll
    for (int i = 0; i < 8; ++i) acc[i] = vzero;

    const int arow = tid >> 1;
    const int acol = (tid & 1) * 16;
    const bool avalid = (m0 + arow < Me);
    const int bk = tid >> 3;
    const int bc = (tid & 7) * 16;
    const size_t wbase = (size_t)e * IDIM * HDIM;
    (void)avalid; (void)arow; (void)acol; (void)rowsRem;

    for (int kt = 0; kt < IDIM / BK; ++kt) {
        const int k0 = kt * BK;
#if HAVE_TDM
        if (tid < 32) {                 // one wave drives the TDM
            tdm_load_a_tile(hbuf + (size_t)(row0 + m0) * IDIM + k0,
                            (unsigned)(uintptr_t)&As[0][0], rowsRem);
        }
#else
        if (avalid) {
            const uint4* src =
                (const uint4*)(hbuf + (size_t)(row0 + m0 + arow) * IDIM + k0 + acol);
            *(uint4*)&As[arow][acol]     = src[0];
            *(uint4*)&As[arow][acol + 8] = src[1];
        } else {
            const uint4 z = {0, 0, 0, 0};
            *(uint4*)&As[arow][acol]     = z;
            *(uint4*)&As[arow][acol + 8] = z;
        }
#endif
        {
            const size_t roff = wbase + (size_t)(k0 + bk) * HDIM + n0 + bc;
            const float4* sb = (const float4*)(wd + roff);
#pragma unroll
            for (int q = 0; q < 4; ++q) {
                float4 v = sb[q];
                Bs[bc + q * 4 + 0][bk] = f2bf(v.x);
                Bs[bc + q * 4 + 1][bk] = f2bf(v.y);
                Bs[bc + q * 4 + 2][bk] = f2bf(v.z);
                Bs[bc + q * 4 + 3][bk] = f2bf(v.w);
            }
            if (kt + 1 < IDIM / BK)
                __builtin_prefetch(wd + roff + (size_t)BK * HDIM, 0, 0);
        }
#if HAVE_TDM
        if (tid < 32) __builtin_amdgcn_s_wait_tensorcnt(0);
#endif
        __syncthreads();
        Frag fa;
        fa.q[0] = *(const uint4*)&As[wave * 16 + ln][half * 8];
        fa.q[1] = *(const uint4*)&As[wave * 16 + ln][16 + half * 8];
        Frag fb;
        fb.q[0] = *(const uint4*)&Bs[ln][half * 8];
        fb.q[1] = *(const uint4*)&Bs[ln][16 + half * 8];
#pragma unroll
        for (int ns = 0; ns < 8; ++ns) {
            Frag fbn;
            if (ns < 7) {
                fbn.q[0] = *(const uint4*)&Bs[(ns + 1) * 16 + ln][half * 8];
                fbn.q[1] = *(const uint4*)&Bs[(ns + 1) * 16 + ln][16 + half * 8];
            } else fbn = fb;
            acc[ns] = __builtin_amdgcn_wmma_f32_16x16x32_bf16(
                false, fa.v, false, fb.v, (short)0, acc[ns], false, false);
            fb = fbn;
        }
        __syncthreads();
    }
#pragma unroll
    for (int ns = 0; ns < 8; ++ns) {
#pragma unroll
        for (int r = 0; r < 8; ++r) {
            const int gm = m0 + wave * 16 + half * 8 + r;
            if (gm < Me)
                outbuf[(size_t)(row0 + gm) * HDIM + n0 + ns * 16 + ln] = acc[ns][r];
        }
    }
}

// ------------------------------------------------- weighted gather-combine
__global__ __launch_bounds__(256)
void combine_kernel(const float* __restrict__ outbuf, const float* __restrict__ topkw,
                    const int* __restrict__ inv, float* __restrict__ y) {
    const int i = blockIdx.x * blockDim.x + threadIdx.x;
    const int HQ = HDIM / 4;
    const int t = i / HQ, hq = i % HQ;
    if (t >= TOK) return;
    float4 acc = {0.f, 0.f, 0.f, 0.f};
#pragma unroll
    for (int k = 0; k < TOPK; ++k) {
        const float w  = topkw[t * TOPK + k];
        const int slot = inv[t * TOPK + k];
        const float4 v = ((const float4*)(outbuf + (size_t)slot * HDIM))[hq];
        acc.x += w * v.x; acc.y += w * v.y; acc.z += w * v.z; acc.w += w * v.w;
    }
    ((float4*)y)[i] = acc;
}

// ----------------------------------------------------------------- launcher
extern "C" void kernel_launch(void* const* d_in, const int* in_sizes, int n_in,
                              void* d_out, int out_size, void* d_ws, size_t ws_size,
                              hipStream_t stream) {
    const float* x  = (const float*)d_in[0];   // hidden_states [B,S,H]
    const float* gw = (const float*)d_in[1];   // gate_weight   [E,H]
    const float* wg = (const float*)d_in[2];   // w_gate        [E,H,I]
    const float* wu = (const float*)d_in[3];   // w_up          [E,H,I]
    const float* wd = (const float*)d_in[4];   // w_down        [E,I,H]
    float* y = (float*)d_out;

    char* p = (char*)d_ws;
    float*          outbuf = (float*)p;          p += (size_t)NROWS * HDIM * sizeof(float);
    unsigned short* hbuf   = (unsigned short*)p; p += (size_t)NROWS * IDIM * sizeof(unsigned short);
    float*          logits = (float*)p;          p += (size_t)TOK * NEXP * sizeof(float);
    float*          topkw  = (float*)p;          p += (size_t)TOK * TOPK * sizeof(float);
    int*            topki  = (int*)p;            p += (size_t)TOK * TOPK * sizeof(int);
    int*            perm   = (int*)p;            p += (size_t)NROWS * sizeof(int);
    int*            inv    = (int*)p;            p += (size_t)NROWS * sizeof(int);
    int*            counts = (int*)p;            p += 64 * sizeof(int);
    int*            offs   = (int*)p;            p += 68 * sizeof(int);
    int*            fill   = (int*)p;            p += 64 * sizeof(int);

    zero_counts<<<1, 64, 0, stream>>>(counts);
    router_gemm<<<TOK / 32, 256, 0, stream>>>(x, gw, logits);
    topk_kernel<<<(TOK + 255) / 256, 256, 0, stream>>>(logits, topkw, topki, counts);
    offsets_kernel<<<1, 1, 0, stream>>>(counts, offs, fill);
    scatter_kernel<<<NROWS / 256, 256, 0, stream>>>(topki, fill, perm, inv);
    gateup_gemm<<<dim3(IDIM / BN, NROWS / BM, NEXP), 256, 0, stream>>>(x, wg, wu, perm, offs, hbuf);
    down_gemm<<<dim3(HDIM / BN, NROWS / BM, NEXP), 256, 0, stream>>>(hbuf, wd, offs, outbuf);
    combine_kernel<<<(TOK * HDIM / 4) / 256, 256, 0, stream>>>(outbuf, topkw, inv, y);
}
